// Model_16690242913042
// MI455X (gfx1250) — compile-verified, matrix operand order checked by reading, not run on hardware
//
#include <hip/hip_runtime.h>

// ---------------------------------------------------------------------------
// MI455X (gfx1250) hetero-GraphSAGE pipeline.
// Precision: bf16 activations, v_wmma_f32_16x16x32_bf16 (fp32 accum),
// fp32 atomic segment-sum. Workload is memory-bound (~2.5 GB @ 23.3 TB/s).
// GEMM B-fragments are fed from LDS; the LDS weight staging uses the Tensor
// Data Mover (tensor_load_to_lds + s_wait_tensorcnt) when available.
// ---------------------------------------------------------------------------

typedef __bf16 bf16;
typedef __attribute__((ext_vector_type(16))) __bf16 v16bf;
typedef __attribute__((ext_vector_type(8)))  __bf16 v8bf;
typedef __attribute__((ext_vector_type(8)))  float  v8f;

typedef unsigned int u32;
typedef __attribute__((ext_vector_type(4))) u32 u32x4;
typedef __attribute__((ext_vector_type(4))) int i32x4;
typedef __attribute__((ext_vector_type(8))) int i32x8;

#define HID 128

#if __has_builtin(__builtin_amdgcn_tensor_load_to_lds) && \
    __has_builtin(__builtin_amdgcn_s_wait_tensorcnt)
#define HAVE_TDM 1
#else
#define HAVE_TDM 0
#endif

// ---- WMMA helper -----------------------------------------------------------
static __device__ inline v8f wmma_bf16(v16bf a, v16bf b, v8f c) {
  return __builtin_amdgcn_wmma_f32_16x16x32_bf16(
      /*neg_a=*/false, a, /*neg_b=*/false, b,
      /*c_mod=*/(short)0, c, /*reuse_a=*/false, /*reuse_b=*/false);
}

// A-fragment (16x32 bf16): lane holds row = lane&15, K chunks [off,off+8) and
// [off+16,off+24) where off = (lane>=16)*8.  Two contiguous 16B loads.
static __device__ inline v16bf load_a_bf(const bf16* p) {
  v8bf lo = *(const v8bf*)p;
  v8bf hi = *(const v8bf*)(p + 16);
  v16bf r;
#pragma unroll
  for (int i = 0; i < 8; ++i) { r[i] = lo[i]; r[i + 8] = hi[i]; }
  return r;
}

// Same fragment but converting from fp32 source on the fly.
static __device__ inline v16bf load_a_f32(const float* p) {
  v16bf r;
#pragma unroll
  for (int i = 0; i < 8; ++i) {
    r[i]     = (bf16)p[i];
    r[i + 8] = (bf16)p[16 + i];
  }
  return r;
}

// B-fragment from packed layout: fragment `frag` = 512 bf16 organized as two
// 512B half-planes (chunk c, lane, 8 values) so each wave32 ds_load_b128 is
// lane-contiguous (minimal LDS bank passes).
static __device__ inline v16bf load_b_frag(const bf16* W, int frag, int lane) {
  const bf16* fb = W + (size_t)frag * 512 + lane * 8;
  v8bf lo = *(const v8bf*)fb;
  v8bf hi = *(const v8bf*)(fb + 256);
  v16bf r;
#pragma unroll
  for (int i = 0; i < 8; ++i) { r[i] = lo[i]; r[i + 8] = hi[i]; }
  return r;
}

// ---- LDS weight staging ----------------------------------------------------
#if HAVE_TDM
// Issue one TDM 1-D tile load: global (nbytes, 8B units) -> LDS.
// D# per CDNA5 ISA ch.8: group0 = {count, lds_addr, global_addr, type=2},
// group1 = {data_size=8B, tensor_dim0=tile_dim0=nbytes/8, dim1=1, stride0}.
// amdgpu-toolchain (clang-23) builtin arity: 6 args, trailing i32x8 + cpol.
static __device__ inline void tdm_issue(bf16* dst, const bf16* src, u32 nbytes) {
  u32 e8 = nbytes >> 3;  // 8-byte elements (fits 16-bit tile_dim0: <= 8192)
  unsigned long long ga = (unsigned long long)(uintptr_t)src;
  u32 lds = (u32)(uintptr_t)dst;  // generic LDS addr: low 32 bits = LDS offset
  u32x4 g0;
  g0[0] = 1u;                                             // count=1 (user D#)
  g0[1] = lds;                                            // lds_addr
  g0[2] = (u32)(ga & 0xFFFFFFFFu);                        // global_addr[31:0]
  g0[3] = (u32)((ga >> 32) & 0x01FFFFFFu) | (2u << 30);   // [56:32] | type=2
  i32x8 g1;
  g1[0] = (int)(3u << 16);                       // wg_mask=0, data_size=8B
  g1[1] = (int)((e8 & 0xFFFFu) << 16);           // tensor_dim0[15:0]
  g1[2] = (int)(((e8 >> 16) & 0xFFFFu) | (1u << 16));  // dim0[31:16], dim1=1
  g1[3] = (int)((e8 & 0xFFFFu) << 16);           // dim1[31:16]=0, tile_dim0
  g1[4] = 1;                                     // tile_dim1=1, tile_dim2=0
  g1[5] = (int)e8;                               // tensor_dim0_stride[31:0]
  g1[6] = 0;
  g1[7] = 0;
  i32x4 z4 = {0, 0, 0, 0};                       // groups 2/3 unused (<=2D)
  i32x8 z8 = {0, 0, 0, 0, 0, 0, 0, 0};
  __builtin_amdgcn_tensor_load_to_lds(g0, g1, z4, z4, z8, 0);
}
#endif

// Cooperative fallback copy, 256 threads, elems multiple of 2048.
static __device__ inline void coop_copy(bf16* dst, const bf16* src, int elems) {
  int chunks = elems >> 3;
  for (int i = threadIdx.x; i < chunks; i += 256)
    ((v8bf*)dst)[i] = ((const v8bf*)src)[i];
}

static __device__ inline void stage_one(bf16* dst, const bf16* src, u32 nbytes) {
#if HAVE_TDM
  if (threadIdx.x < 32) {          // one wave drives the DMA
    tdm_issue(dst, src, nbytes);
    __builtin_amdgcn_s_wait_tensorcnt(0);
  }
#else
  coop_copy(dst, src, (int)(nbytes >> 1));
#endif
  __syncthreads();
}

static __device__ inline void stage_two(bf16* d0, const bf16* s0, u32 n0,
                                        bf16* d1, const bf16* s1, u32 n1) {
#if HAVE_TDM
  if (threadIdx.x < 32) {
    tdm_issue(d0, s0, n0);
    tdm_issue(d1, s1, n1);
    __builtin_amdgcn_s_wait_tensorcnt(0);
  }
#else
  coop_copy(d0, s0, (int)(n0 >> 1));
  coop_copy(d1, s1, (int)(n1 >> 1));
#endif
  __syncthreads();
}

// ---- weight pre-pack -------------------------------------------------------
// Pack row-major fp32 W[K,128] into per-lane B-fragment planes:
// frag = kt*8+nt; packed[frag*512 + c*256 + lane*8 + i]
//   = bf16( W[(kt*32 + (c?i+16:i) + (lane>=16?8:0)) * 128 + nt*16 + (lane&15)] )
__global__ __launch_bounds__(256) void k_pack_w(const float* __restrict__ W,
                                                bf16* __restrict__ out, int K) {
  int tid = blockIdx.x * 256 + threadIdx.x;
  int total = (K / 32) * 4096;
  if (tid >= total) return;
  int i    = tid & 7;
  int lane = (tid >> 3) & 31;
  int c    = (tid >> 8) & 1;
  int nt   = (tid >> 9) & 7;
  int kt   = tid >> 12;
  int k = kt * 32 + (c ? i + 16 : i) + ((lane >= 16) ? 8 : 0);
  int n = nt * 16 + (lane & 15);
  out[tid] = (bf16)W[(size_t)k * HID + n];
}

// ---- fused input: out = bf16( X@W + bias + emb[ids[row]] ), K=256 ----------
__global__ __launch_bounds__(256) void k_gemm_input(
    const float* __restrict__ X, const bf16* __restrict__ Wpk,
    const float* __restrict__ bias, const float* __restrict__ emb,
    const int* __restrict__ ids, bf16* __restrict__ out, int M) {
  __shared__ bf16 sW[32768];  // 64 KB: full K=256 packed weight
  stage_one(sW, Wpk, 65536);

  const int wave = threadIdx.x >> 5, lane = threadIdx.x & 31;
  const int laneLo = lane & 15, hiOff = (lane >> 4) * 8;
  const int rowBase = blockIdx.x * 128 + wave * 16;
  int arow = rowBase + laneLo; if (arow > M - 1) arow = M - 1;

  v8f acc[8] = {};
  const float* abase = X + (size_t)arow * 256 + hiOff;
#pragma unroll
  for (int kt = 0; kt < 8; ++kt) {
    v16bf a = load_a_f32(abase + kt * 32);
    v16bf b = load_b_frag(sW, kt * 8, lane);
#pragma unroll
    for (int nt = 0; nt < 8; ++nt) {
      v16bf bn;
      if (nt < 7) bn = load_b_frag(sW, kt * 8 + nt + 1, lane);
      acc[nt] = wmma_bf16(a, b, acc[nt]);
      b = bn;
    }
  }
  const int rhi = rowBase + (lane >> 4) * 8;
#pragma unroll
  for (int v = 0; v < 8; ++v) {
    int row = rhi + v;
    if (row < M) {
      int id = ids[row];
#pragma unroll
      for (int nt = 0; nt < 8; ++nt) {
        int col = nt * 16 + laneLo;
        float val = acc[nt][v] + bias[col] + emb[(size_t)id * HID + col];
        out[(size_t)row * HID + col] = (bf16)val;
      }
    }
  }
}

// ---- SAGE combine: out = (relu)( A1@Wl + bias + A2@Wr ), K=128 each --------
__global__ __launch_bounds__(256) void k_gemm_sage(
    const bf16* __restrict__ A1, const bf16* __restrict__ A2,
    const bf16* __restrict__ Wl, const bf16* __restrict__ Wr,
    const float* __restrict__ bias, bf16* __restrict__ out, int M, int relu) {
  __shared__ bf16 sW[32768];  // 64 KB: Wl at 0, Wr at +16384
  stage_two(sW, Wl, 32768, sW + 16384, Wr, 32768);

  const int wave = threadIdx.x >> 5, lane = threadIdx.x & 31;
  const int laneLo = lane & 15, hiOff = (lane >> 4) * 8;
  const int rowBase = blockIdx.x * 128 + wave * 16;
  int arow = rowBase + laneLo; if (arow > M - 1) arow = M - 1;

  v8f acc[8] = {};
  const bf16* a1 = A1 + (size_t)arow * HID + hiOff;
  const bf16* a2 = A2 + (size_t)arow * HID + hiOff;
#pragma unroll
  for (int kt = 0; kt < 4; ++kt) {
    v16bf a = load_a_bf(a1 + kt * 32);
    v16bf b = load_b_frag(sW, kt * 8, lane);
#pragma unroll
    for (int nt = 0; nt < 8; ++nt) {
      v16bf bn;
      if (nt < 7) bn = load_b_frag(sW, kt * 8 + nt + 1, lane);
      acc[nt] = wmma_bf16(a, b, acc[nt]);
      b = bn;
    }
  }
#pragma unroll
  for (int kt = 0; kt < 4; ++kt) {
    v16bf a = load_a_bf(a2 + kt * 32);
    v16bf b = load_b_frag(sW + 16384, kt * 8, lane);
#pragma unroll
    for (int nt = 0; nt < 8; ++nt) {
      v16bf bn;
      if (nt < 7) bn = load_b_frag(sW + 16384, kt * 8 + nt + 1, lane);
      acc[nt] = wmma_bf16(a, b, acc[nt]);
      b = bn;
    }
  }
  const int rhi = rowBase + (lane >> 4) * 8;
#pragma unroll
  for (int v = 0; v < 8; ++v) {
    int row = rhi + v;
    if (row < M) {
#pragma unroll
      for (int nt = 0; nt < 8; ++nt) {
        int col = nt * 16 + laneLo;
        float val = acc[nt][v] + bias[col];
        if (relu) val = fmaxf(val, 0.0f);
        out[(size_t)row * HID + col] = (bf16)val;
      }
    }
  }
}

// ---- classifier GEMM: h = relu( [o_p[lp]||o_s[ls]] @ Wc1 + bc1 ), K=256 ----
__global__ __launch_bounds__(256) void k_gemm_cls(
    const bf16* __restrict__ op, const bf16* __restrict__ os,
    const int* __restrict__ lp, const int* __restrict__ ls,
    const bf16* __restrict__ Wpk, const float* __restrict__ bias,
    bf16* __restrict__ out, int M) {
  __shared__ bf16 sW[32768];  // 64 KB: full K=256 packed Wc1
  stage_one(sW, Wpk, 65536);

  const int wave = threadIdx.x >> 5, lane = threadIdx.x & 31;
  const int laneLo = lane & 15, hiOff = (lane >> 4) * 8;
  const int rowBase = blockIdx.x * 128 + wave * 16;
  int arow = rowBase + laneLo; if (arow > M - 1) arow = M - 1;
  int rp = lp[arow], rs = ls[arow];

  v8f acc[8] = {};
  const bf16* a1 = op + (size_t)rp * HID + hiOff;
  const bf16* a2 = os + (size_t)rs * HID + hiOff;
#pragma unroll
  for (int kt = 0; kt < 4; ++kt) {
    v16bf a = load_a_bf(a1 + kt * 32);
    v16bf b = load_b_frag(sW, kt * 8, lane);
#pragma unroll
    for (int nt = 0; nt < 8; ++nt) {
      v16bf bn;
      if (nt < 7) bn = load_b_frag(sW, kt * 8 + nt + 1, lane);
      acc[nt] = wmma_bf16(a, b, acc[nt]);
      b = bn;
    }
  }
#pragma unroll
  for (int kt = 0; kt < 4; ++kt) {
    v16bf a = load_a_bf(a2 + kt * 32);
    v16bf b = load_b_frag(sW, (kt + 4) * 8, lane);
#pragma unroll
    for (int nt = 0; nt < 8; ++nt) {
      v16bf bn;
      if (nt < 7) bn = load_b_frag(sW, (kt + 4) * 8 + nt + 1, lane);
      acc[nt] = wmma_bf16(a, b, acc[nt]);
      b = bn;
    }
  }
  const int rhi = rowBase + (lane >> 4) * 8;
#pragma unroll
  for (int v = 0; v < 8; ++v) {
    int row = rhi + v;
    if (row < M) {
#pragma unroll
      for (int nt = 0; nt < 8; ++nt) {
        int col = nt * 16 + laneLo;
        float val = fmaxf(acc[nt][v] + bias[col], 0.0f);
        out[(size_t)row * HID + col] = (bf16)val;
      }
    }
  }
}

// ---- edge scatter: agg[dst] += x[src] (fp32 atomics, bf16 source) ----------
__global__ __launch_bounds__(256) void k_scatter(
    const bf16* __restrict__ x, const int* __restrict__ gi,
    const int* __restrict__ si, int E, float* __restrict__ agg) {
  long long tid = (long long)blockIdx.x * 256 + threadIdx.x;
  if (tid >= (long long)E * HID) return;
  int e = (int)(tid >> 7), f = (int)(tid & 127);
  int s = gi[e], d = si[e];
  float v = (float)x[(size_t)s * HID + f];
  unsafeAtomicAdd(&agg[(size_t)d * HID + f], v);
}

__global__ __launch_bounds__(256) void k_deg(const int* __restrict__ idx, int E,
                                             float* __restrict__ deg) {
  int e = blockIdx.x * 256 + threadIdx.x;
  if (e < E) unsafeAtomicAdd(&deg[idx[e]], 1.0f);
}

// ---- normalize by degree + bf16 convert ------------------------------------
__global__ __launch_bounds__(256) void k_norm(const float* __restrict__ agg,
                                              const float* __restrict__ deg,
                                              bf16* __restrict__ out, int n) {
  int tid = blockIdx.x * 256 + threadIdx.x;
  if (tid >= n * HID) return;
  int i = tid >> 7;
  float r = 1.0f / fmaxf(deg[i], 1.0f);
  out[tid] = (bf16)(agg[tid] * r);
}

// ---- final per-row dot: out[l] = h[l,:]·Wc2 + bc2 --------------------------
__global__ __launch_bounds__(256) void k_dot(const bf16* __restrict__ h,
                                             const float* __restrict__ w,
                                             const float* __restrict__ b,
                                             float* __restrict__ out, int L) {
  int l = blockIdx.x * 256 + threadIdx.x;
  if (l >= L) return;
  const bf16* p = h + (size_t)l * HID;
  float s = 0.0f;
#pragma unroll
  for (int f0 = 0; f0 < HID; f0 += 8) {
    v8bf v = *(const v8bf*)(p + f0);
#pragma unroll
    for (int i = 0; i < 8; ++i) s += (float)v[i] * w[f0 + i];
  }
  out[l] = s + b[0];
}

// ---------------------------------------------------------------------------
extern "C" void kernel_launch(void* const* d_in, const int* in_sizes, int n_in,
                              void* d_out, int out_size, void* d_ws, size_t ws_size,
                              hipStream_t stream) {
  (void)n_in; (void)out_size; (void)ws_size;
  const float* x_paper      = (const float*)d_in[0];
  const float* x_software   = (const float*)d_in[1];
  const int*   paper_id     = (const int*)d_in[2];
  const int*   software_id  = (const int*)d_in[3];
  const int*   edge_src     = (const int*)d_in[4];
  const int*   edge_dst     = (const int*)d_in[5];
  const int*   lbl_paper    = (const int*)d_in[6];
  const int*   lbl_software = (const int*)d_in[7];
  const float* paper_emb    = (const float*)d_in[8];
  const float* software_emb = (const float*)d_in[9];
  const float* Wp  = (const float*)d_in[10]; const float* bp  = (const float*)d_in[11];
  const float* Ws  = (const float*)d_in[12]; const float* bs  = (const float*)d_in[13];
  const float* Wl1_ps = (const float*)d_in[14]; const float* bl1_ps = (const float*)d_in[15];
  const float* Wr1_ps = (const float*)d_in[16];
  const float* Wl1_sp = (const float*)d_in[17]; const float* bl1_sp = (const float*)d_in[18];
  const float* Wr1_sp = (const float*)d_in[19];
  const float* Wl2_ps = (const float*)d_in[20]; const float* bl2_ps = (const float*)d_in[21];
  const float* Wr2_ps = (const float*)d_in[22];
  const float* Wl2_sp = (const float*)d_in[23]; const float* bl2_sp = (const float*)d_in[24];
  const float* Wr2_sp = (const float*)d_in[25];
  const float* Wc1 = (const float*)d_in[26]; const float* bc1 = (const float*)d_in[27];
  const float* Wc2 = (const float*)d_in[28]; const float* bc2 = (const float*)d_in[29];
  float* out = (float*)d_out;

  const int NP = in_sizes[2], NS = in_sizes[3], E = in_sizes[4], L = in_sizes[6];

  // ---- workspace carving (256B aligned) ----
  char* base = (char*)d_ws;
  size_t off = 0;
  auto alloc = [&](size_t bytes) -> char* {
    char* p = base + off;
    off = (off + bytes + 255) & ~(size_t)255;
    return p;
  };
  bf16* xp_bf   = (bf16*)alloc((size_t)NP * HID * 2);
  bf16* xs_bf   = (bf16*)alloc((size_t)NS * HID * 2);
  bf16* hp_bf   = (bf16*)alloc((size_t)NP * HID * 2);
  bf16* hs_bf   = (bf16*)alloc((size_t)NS * HID * 2);
  bf16* op_bf   = (bf16*)alloc((size_t)NP * HID * 2);
  bf16* os_bf   = (bf16*)alloc((size_t)NS * HID * 2);
  bf16* aggp_bf = (bf16*)alloc((size_t)NP * HID * 2);
  bf16* aggs_bf = (bf16*)alloc((size_t)NS * HID * 2);
  bf16* hcls_bf = (bf16*)alloc((size_t)L * HID * 2);
  float* agg_p  = (float*)alloc((size_t)NP * HID * 4);
  float* agg_s  = (float*)alloc((size_t)NS * HID * 4);
  float* deg_p  = (float*)alloc((size_t)NP * 4);
  float* deg_s  = (float*)alloc((size_t)NS * 4);
  bf16* wp_pk   = (bf16*)alloc((size_t)256 * HID * 2);
  bf16* ws_pk   = (bf16*)alloc((size_t)256 * HID * 2);
  bf16* wc1_pk  = (bf16*)alloc((size_t)256 * HID * 2);
  bf16* wl1ps_pk = (bf16*)alloc((size_t)128 * HID * 2);
  bf16* wr1ps_pk = (bf16*)alloc((size_t)128 * HID * 2);
  bf16* wl1sp_pk = (bf16*)alloc((size_t)128 * HID * 2);
  bf16* wr1sp_pk = (bf16*)alloc((size_t)128 * HID * 2);
  bf16* wl2ps_pk = (bf16*)alloc((size_t)128 * HID * 2);
  bf16* wr2ps_pk = (bf16*)alloc((size_t)128 * HID * 2);
  bf16* wl2sp_pk = (bf16*)alloc((size_t)128 * HID * 2);
  bf16* wr2sp_pk = (bf16*)alloc((size_t)128 * HID * 2);

  // ---- weight packing ----
  auto pack = [&](const float* W, bf16* pk, int K) {
    int total = (K / 32) * 4096;
    k_pack_w<<<(total + 255) / 256, 256, 0, stream>>>(W, pk, K);
  };
  pack(Wp, wp_pk, 256);       pack(Ws, ws_pk, 256);     pack(Wc1, wc1_pk, 256);
  pack(Wl1_ps, wl1ps_pk, 128); pack(Wr1_ps, wr1ps_pk, 128);
  pack(Wl1_sp, wl1sp_pk, 128); pack(Wr1_sp, wr1sp_pk, 128);
  pack(Wl2_ps, wl2ps_pk, 128); pack(Wr2_ps, wr2ps_pk, 128);
  pack(Wl2_sp, wl2sp_pk, 128); pack(Wr2_sp, wr2sp_pk, 128);

  // ---- input fusion ----
  k_gemm_input<<<(NP + 127) / 128, 256, 0, stream>>>(x_paper, wp_pk, bp,
                                                     paper_emb, paper_id, xp_bf, NP);
  k_gemm_input<<<(NS + 127) / 128, 256, 0, stream>>>(x_software, ws_pk, bs,
                                                     software_emb, software_id, xs_bf, NS);

  // ---- degrees (edge structure is layer-invariant) ----
  (void)hipMemsetAsync(deg_p, 0, (size_t)NP * 4, stream);
  (void)hipMemsetAsync(deg_s, 0, (size_t)NS * 4, stream);
  k_deg<<<(E + 255) / 256, 256, 0, stream>>>(edge_dst, E, deg_s);
  k_deg<<<(E + 255) / 256, 256, 0, stream>>>(edge_src, E, deg_p);

  const int scat_blocks = (int)(((long long)E * HID + 255) / 256);

  // ---- layer 1: software dst (paper -> software) ----
  (void)hipMemsetAsync(agg_s, 0, (size_t)NS * HID * 4, stream);
  k_scatter<<<scat_blocks, 256, 0, stream>>>(xp_bf, edge_src, edge_dst, E, agg_s);
  k_norm<<<(NS * HID + 255) / 256, 256, 0, stream>>>(agg_s, deg_s, aggs_bf, NS);
  k_gemm_sage<<<(NS + 127) / 128, 256, 0, stream>>>(aggs_bf, xs_bf, wl1ps_pk,
                                                    wr1ps_pk, bl1_ps, hs_bf, NS, 1);

  // ---- layer 1: paper dst (software -> paper) ----
  (void)hipMemsetAsync(agg_p, 0, (size_t)NP * HID * 4, stream);
  k_scatter<<<scat_blocks, 256, 0, stream>>>(xs_bf, edge_dst, edge_src, E, agg_p);
  k_norm<<<(NP * HID + 255) / 256, 256, 0, stream>>>(agg_p, deg_p, aggp_bf, NP);
  k_gemm_sage<<<(NP + 127) / 128, 256, 0, stream>>>(aggp_bf, xp_bf, wl1sp_pk,
                                                    wr1sp_pk, bl1_sp, hp_bf, NP, 1);

  // ---- layer 2: software dst ----
  (void)hipMemsetAsync(agg_s, 0, (size_t)NS * HID * 4, stream);
  k_scatter<<<scat_blocks, 256, 0, stream>>>(hp_bf, edge_src, edge_dst, E, agg_s);
  k_norm<<<(NS * HID + 255) / 256, 256, 0, stream>>>(agg_s, deg_s, aggs_bf, NS);
  k_gemm_sage<<<(NS + 127) / 128, 256, 0, stream>>>(aggs_bf, hs_bf, wl2ps_pk,
                                                    wr2ps_pk, bl2_ps, os_bf, NS, 0);

  // ---- layer 2: paper dst ----
  (void)hipMemsetAsync(agg_p, 0, (size_t)NP * HID * 4, stream);
  k_scatter<<<scat_blocks, 256, 0, stream>>>(hs_bf, edge_dst, edge_src, E, agg_p);
  k_norm<<<(NP * HID + 255) / 256, 256, 0, stream>>>(agg_p, deg_p, aggp_bf, NP);
  k_gemm_sage<<<(NP + 127) / 128, 256, 0, stream>>>(aggp_bf, hp_bf, wl2sp_pk,
                                                    wr2sp_pk, bl2_sp, op_bf, NP, 0);

  // ---- edge classifier ----
  k_gemm_cls<<<(L + 127) / 128, 256, 0, stream>>>(op_bf, os_bf, lbl_paper,
                                                  lbl_software, wc1_pk, bc1, hcls_bf, L);
  k_dot<<<(L + 255) / 256, 256, 0, stream>>>(hcls_bf, Wc2, bc2, out, L);
}
